// SBVisualCore_7224134991879
// MI455X (gfx1250) — matrix operand order repaired and checked by reading, not compiled, not run on hardware
//
#include <hip/hip_runtime.h>
#include <math.h>

// ---- problem constants ----
#define B_  512
#define N_  196
#define D_  128
#define K_  4
#define ITILES 49          // N_/4 i-tiles per batch
#define TILES  (B_*ITILES) // 25088 16-row tiles
#define WAVES  4

typedef __attribute__((ext_vector_type(16))) __bf16 v16bf;
typedef __attribute__((ext_vector_type(8)))  float  v8f;

// d_out layout (floats): [relation_state | neighbor_index(as float) | dist_mean]
#define OUT_NBR   ((size_t)B_*N_*D_)           // 12,845,056
#define OUT_DMEAN (OUT_NBR + (size_t)B_*N_*K_) // +401,408

// d_ws layout (bytes):
//   Wt1 bf16 128x256 | Wt2 bf16 128x128 | nidx int32 B*N*K | signals bf16 B*N*D
#define WS_WT1  0
#define WS_WT2  65536
#define WS_NIDX 98304
#define WS_SIGB (98304 + (size_t)B_*N_*K_*4)   // 1,703,936
#define NSIG    ((size_t)B_*N_*D_)             // 12,845,056 elements

// ---------------------------------------------------------------------------
// Kernel 0a: transpose + convert weights to bf16, column-major (Wt[n][k])
// so WMMA B-fragments are contiguous 32B per lane.
// ---------------------------------------------------------------------------
__global__ __launch_bounds__(256) void prep_weights(const float* __restrict__ W1,
                                                    const float* __restrict__ W2,
                                                    __bf16* __restrict__ Wt1,
                                                    __bf16* __restrict__ Wt2) {
  int t = blockIdx.x * 256 + threadIdx.x;
  if (t < 128 * 256) {                // Wt1[n][k] = W1[k][n], k<256
    int n = t >> 8, k = t & 255;
    Wt1[t] = (__bf16)W1[(size_t)k * D_ + n];
  } else {
    int u = t - 128 * 256;
    if (u < 128 * 128) {              // Wt2[n][k] = W2[k][n]
      int n = u >> 7, k = u & 127;
      Wt2[u] = (__bf16)W2[(size_t)k * D_ + n];
    }
  }
}

// ---------------------------------------------------------------------------
// Kernel 0b: one-time f32 -> bf16 conversion of the signals array so the hot
// MLP kernel gathers half the bytes and does no per-tile conversions.
// ---------------------------------------------------------------------------
__global__ __launch_bounds__(256) void conv_signals(const float* __restrict__ s,
                                                    __bf16* __restrict__ d) {
  size_t i = ((size_t)blockIdx.x * 256 + threadIdx.x) * 4;  // NSIG % 4 == 0
  if (i < NSIG) {
    float4 v = *(const float4*)(s + i);
    d[i + 0] = (__bf16)v.x;
    d[i + 1] = (__bf16)v.y;
    d[i + 2] = (__bf16)v.z;
    d[i + 3] = (__bf16)v.w;
  }
}

// ---------------------------------------------------------------------------
// Kernel 1: per-batch pairwise distances, top-5 (drop self), distance mean.
// One block (256 threads) per batch; positions staged in LDS.
// ---------------------------------------------------------------------------
__global__ __launch_bounds__(256) void knn_kernel(const float* __restrict__ positions,
                                                  int* __restrict__ nidx_ws,
                                                  float* __restrict__ out_nbr_f,
                                                  float* __restrict__ out_dmean) {
  __shared__ float px[N_], py[N_];
  __shared__ float red[256];
  int b = blockIdx.x, t = threadIdx.x;
  const float* posB = positions + (size_t)b * N_ * 2;
  if (t < N_) { px[t] = posB[t * 2 + 0]; py[t] = posB[t * 2 + 1]; }
  __syncthreads();

  float sum = 0.0f;
  if (t < N_) {
    float mx = px[t], my = py[t];
    float bd[5]; int bi[5];
    #pragma unroll
    for (int k = 0; k < 5; ++k) { bd[k] = 3.4e38f; bi[k] = 0; }
    for (int j = 0; j < N_; ++j) {
      float dx = px[j] - mx, dy = py[j] - my;
      float d = sqrtf(dx * dx + dy * dy);
      sum += d;
      if (d < bd[4]) {                 // strict < keeps lower index first on ties
        int k = 4;
        while (k > 0 && d < bd[k - 1]) { bd[k] = bd[k - 1]; bi[k] = bi[k - 1]; --k; }
        bd[k] = d; bi[k] = j;
      }
    }
    size_t base = ((size_t)b * N_ + t) * K_;
    #pragma unroll
    for (int k = 0; k < K_; ++k) {     // entry 0 is self (d==0): drop it
      nidx_ws[base + k]   = bi[k + 1];
      out_nbr_f[base + k] = (float)bi[k + 1];
    }
  }
  red[t] = sum;
  __syncthreads();
  for (int s = 128; s > 0; s >>= 1) {
    if (t < s) red[t] += red[t + s];
    __syncthreads();
  }
  if (t == 0) out_dmean[b] = red[0] * (1.0f / ((float)N_ * (float)N_));
}

// ---------------------------------------------------------------------------
// Kernel 2: gathered MLP via bf16 WMMA. One wave per 16-row tile
// (rows = 4 consecutive i's x 4 neighbor slots), 4 waves per block.
// GEMM1: 16x256 @ 256x128 (+rank-2 relpos update) -> GELU
// GEMM2: 16x128 @ 128x128 -> mean over the 4 neighbor slots per i.
// ---------------------------------------------------------------------------
__global__ __launch_bounds__(128) void mlp_kernel(const __bf16* __restrict__ sigb,  // pre-converted bf16
                                                  const float* __restrict__ positions,
                                                  const float* __restrict__ W1,   // f32, rows 256/257 used
                                                  const float* __restrict__ b1,
                                                  const float* __restrict__ b2,
                                                  const __bf16* __restrict__ Wt1, // 128x256 col-major
                                                  const __bf16* __restrict__ Wt2, // 128x128 col-major
                                                  const int* __restrict__ nidx,
                                                  float* __restrict__ rstate) {
  __shared__ int    ldsNI[WAVES][16];
  __shared__ float  ldsRPx[WAVES][16];
  __shared__ float  ldsRPy[WAVES][16];
  __shared__ __bf16 ldsH[WAVES][16][D_];   // hidden, row-major bf16

  int lane = threadIdx.x & 31;
  int wv   = threadIdx.x >> 5;
  int tile = blockIdx.x * WAVES + wv;      // grid sized exactly: no guard
  int b  = tile / ITILES;
  int i0 = (tile % ITILES) * 4;

  const float*  posB = positions + (size_t)b * N_ * 2;
  const __bf16* sigB = sigb      + (size_t)b * N_ * D_;

  if (lane < 16) {                         // stage neighbor idx + rel positions
    int r = lane;
    int i = i0 + (r >> 2);
    int ni = nidx[((size_t)b * N_ + i) * K_ + (r & 3)];
    ldsNI[wv][r]  = ni;
    ldsRPx[wv][r] = posB[ni * 2 + 0] - posB[i * 2 + 0];
    ldsRPy[wv][r] = posB[ni * 2 + 1] - posB[i * 2 + 1];
  }
  __syncthreads();

  int r  = lane & 15;                      // matrix row this lane owns
  int hi = lane >> 4;                      // lane half selects K sub-range
  int c  = hi ? 16 : 0;
  const __bf16* srcRow = sigB + (size_t)(i0 + (r >> 2)) * D_;
  const __bf16* nbrRow = sigB + (size_t)ldsNI[wv][r] * D_;

  // A fragments for GEMM1: K=256 = [source 0..127 | neighbor 0..127].
  // K order chosen here (matches B): lane<16 holds K{0-15}, lane>=16 K{16-31}.
  // Each half-fragment is 8 contiguous bf16 (16B, aligned) -> b128-class loads.
  v16bf afrag[8];
  #pragma unroll
  for (int kb = 0; kb < 8; ++kb) {
    const __bf16* rowp = (kb < 4) ? srcRow : nbrRow;
    int base = (kb & 3) * 32 + c;
    #pragma unroll
    for (int t = 0; t < 8; ++t) afrag[kb][t]     = rowp[base + t];
    #pragma unroll
    for (int t = 0; t < 8; ++t) afrag[kb][8 + t] = rowp[base + 8 + t];
  }

  const float gsc = 0.7071067811865476f;
  #pragma unroll
  for (int nt = 0; nt < 8; ++nt) {         // GEMM1 over 8 N-tiles
    int n = nt * 16 + r;                   // output column this lane owns
    float bias = b1[n];
    v8f acc;
    #pragma unroll
    for (int q = 0; q < 8; ++q) acc[q] = bias;
    const __bf16* wcol = Wt1 + (size_t)n * 256 + hi * 16;
    #pragma unroll
    for (int kb = 0; kb < 8; ++kb) {
      v16bf bfrag = *(const v16bf*)(wcol + kb * 32);
      acc = __builtin_amdgcn_wmma_f32_16x16x32_bf16(false, afrag[kb],
                                                    false, bfrag,
                                                    (short)0, acc, false, false);
    }
    // rank-2 relpos update (W1 rows 256,257) + exact GELU, stage hidden to LDS
    float w256 = W1[256 * D_ + n];
    float w257 = W1[257 * D_ + n];
    #pragma unroll
    for (int v = 0; v < 8; ++v) {
      int row = v + hi * 8;                // C layout: VGPR v -> row v (+8 for hi half)
      float pre = acc[v] + ldsRPx[wv][row] * w256 + ldsRPy[wv][row] * w257;
      float h = 0.5f * pre * (1.0f + erff(pre * gsc));
      ldsH[wv][row][n] = (__bf16)h;
    }
  }
  __syncthreads();

  // A fragments for GEMM2 from staged hidden (K=128)
  v16bf a2[4];
  #pragma unroll
  for (int kb = 0; kb < 4; ++kb) {
    int base = kb * 32 + c;
    #pragma unroll
    for (int t = 0; t < 8; ++t) a2[kb][t]     = ldsH[wv][r][base + t];
    #pragma unroll
    for (int t = 0; t < 8; ++t) a2[kb][8 + t] = ldsH[wv][r][base + 8 + t];
  }

  float* outB = rstate + (size_t)b * N_ * D_;
  #pragma unroll
  for (int nt = 0; nt < 8; ++nt) {         // GEMM2 over 8 N-tiles + K-mean epilogue
    int n = nt * 16 + r;
    float bias2 = b2[n];
    v8f acc;
    #pragma unroll
    for (int q = 0; q < 8; ++q) acc[q] = bias2;
    const __bf16* wcol = Wt2 + (size_t)n * 128 + hi * 16;
    #pragma unroll
    for (int kb = 0; kb < 4; ++kb) {
      v16bf bfrag = *(const v16bf*)(wcol + kb * 32);
      acc = __builtin_amdgcn_wmma_f32_16x16x32_bf16(false, a2[kb],
                                                    false, bfrag,
                                                    (short)0, acc, false, false);
    }
    // rows v0..3 = neighbor slots of i_local {0 or 2}; v4..7 = i_local {1 or 3}
    float sLo = (acc[0] + acc[1] + acc[2] + acc[3]) * 0.25f;
    float sHi = (acc[4] + acc[5] + acc[6] + acc[7]) * 0.25f;
    int iA = i0 + hi * 2;
    outB[(size_t)(iA + 0) * D_ + n] = sLo;
    outB[(size_t)(iA + 1) * D_ + n] = sHi;
  }
}

// ---------------------------------------------------------------------------
extern "C" void kernel_launch(void* const* d_in, const int* in_sizes, int n_in,
                              void* d_out, int out_size, void* d_ws, size_t ws_size,
                              hipStream_t stream) {
  const float* signals   = (const float*)d_in[0];
  const float* positions = (const float*)d_in[1];
  const float* W1        = (const float*)d_in[2];
  const float* b1        = (const float*)d_in[3];
  const float* W2        = (const float*)d_in[4];
  const float* b2        = (const float*)d_in[5];

  float* out = (float*)d_out;
  float* out_state = out;
  float* out_nbr   = out + OUT_NBR;
  float* out_dmean = out + OUT_DMEAN;

  char* ws = (char*)d_ws;
  __bf16* Wt1  = (__bf16*)(ws + WS_WT1);
  __bf16* Wt2  = (__bf16*)(ws + WS_WT2);
  int*    nid  = (int*)   (ws + WS_NIDX);
  __bf16* Sigb = (__bf16*)(ws + WS_SIGB);

  prep_weights<<<192, 256, 0, stream>>>(W1, W2, Wt1, Wt2);
  conv_signals<<<(int)(NSIG / 4 / 256), 256, 0, stream>>>(signals, Sigb);
  knn_kernel<<<B_, 256, 0, stream>>>(positions, nid, out_nbr, out_dmean);
  mlp_kernel<<<TILES / WAVES, 32 * WAVES, 0, stream>>>(Sigb, positions, W1, b1, b2,
                                                       Wt1, Wt2, nid, out_state);
}
